// MyNAMBase_40372692583145
// MI455X (gfx1250) — matrix operand order repaired and checked by reading, not compiled, not run on hardware
//
#include <hip/hip_runtime.h>

typedef __attribute__((ext_vector_type(16))) _Float16 v16h;
typedef __attribute__((ext_vector_type(8)))  _Float16 v8h;
typedef __attribute__((ext_vector_type(8)))  float    v8f;

#define B_TOT 8192
#define F_TOT 128
#define H1 64
#define H2 64
#define H3 32
#define CHUNKS 8
#define ROWS_PER_CHUNK (B_TOT / CHUNKS)        // 1024
#define WAVES 8
#define TILE_ROWS 16
#define ROWS_PER_ITER (WAVES * TILE_ROWS)      // 128
#define ITERS (ROWS_PER_CHUNK / ROWS_PER_ITER) // 8
#define H2PAD 72   // padded h2 row stride in halves: 144B -> 16 distinct bank groups

// K index inside a 16x32 f16 A-operand for lane-half h, packed element e
// (contiguous order: k = 16h + e, matching the B operands)
__device__ __forceinline__ int a_k_of(int e, int h) {
    int j = e >> 1;
    int base = h * 16 + 2 * j;
    return base + (e & 1);
}

__global__ void nam_init(float* __restrict__ out, const float* __restrict__ final_bias) {
    int i = blockIdx.x * blockDim.x + threadIdx.x;
    if (i < B_TOT) out[i] = final_bias[0];
}

__global__ __launch_bounds__(256, 1) void nam_fused(
    const float* __restrict__ X,  const float* __restrict__ W1,
    const float* __restrict__ b1, const float* __restrict__ W2,
    const float* __restrict__ b2, const float* __restrict__ W3,
    const float* __restrict__ b3, const float* __restrict__ W4,
    const float* __restrict__ b4, float* __restrict__ out)
{
    // Weights staged in WMMA *A-operand* layout (lane l, elem e):
    //   M = mt*16 + (l&15) = output unit o, K = kt*32 + a_k_of(e, l>>4) = input i
    __shared__ alignas(32) _Float16 sW2A[8][512];  // op = kt*4 + mt
    __shared__ alignas(32) _Float16 sW3A[4][512];  // op = kt*2 + mt
    __shared__ float sEW[H1];                      // exp(W1[f]) — plain linear
    __shared__ float sB2[H2];
    __shared__ float sB3[H3];
    __shared__ float sW4v[H3];
    __shared__ alignas(32) _Float16 sH2[WAVES][TILE_ROWS * H2PAD]; // h2[batch][o], padded

    const int f    = blockIdx.y;
    const int tid  = threadIdx.x;
    const int lane = tid & 31;
    const int w    = tid >> 5;
    const int m    = lane & 15;   // batch lane (B/C column)
    const int h    = lane >> 4;   // lane half

    // ---- stage per-feature weights into LDS (f32 -> f16, A-operand layout) ----
    for (int idx = tid; idx < 8 * 512; idx += 256) {
        int op = idx >> 9, l = (idx >> 4) & 31, e = idx & 15;
        int kt = op >> 2, mt = op & 3;
        int o = mt * 16 + (l & 15);
        int i = kt * 32 + a_k_of(e, l >> 4);
        sW2A[op][(l << 4) + e] = (_Float16)W2[f * H2 * H1 + o * H1 + i]; // W2[f][o][i]
    }
    for (int idx = tid; idx < 4 * 512; idx += 256) {
        int op = idx >> 9, l = (idx >> 4) & 31, e = idx & 15;
        int kt = op >> 1, mt = op & 1;
        int o = mt * 16 + (l & 15);
        int i = kt * 32 + a_k_of(e, l >> 4);
        sW3A[op][(l << 4) + e] = (_Float16)W3[f * H3 * H2 + o * H2 + i]; // W3[f][o][i]
    }
    if (tid < H1) sEW[tid] = expf(W1[f * H1 + tid]);  // W1[f][i][0]
    if (tid < H2) sB2[tid] = b2[f * H2 + tid];
    if (tid < H3) { sB3[tid] = b3[f * H3 + tid]; sW4v[tid] = W4[f * H3 + tid]; }
    __syncthreads();

    // ---- per-wave persistent operands ----
    v16h Aw2[8], Aw3[4];
    #pragma unroll
    for (int op = 0; op < 8; ++op) Aw2[op] = *(const v16h*)&sW2A[op][lane << 4];
    #pragma unroll
    for (int op = 0; op < 4; ++op) Aw3[op] = *(const v16h*)&sW3A[op][lane << 4];

    // EXU factors for this lane's B-operand slots: i = kt*32 + h*16 + e (contiguous)
    float ewB[2][16];
    #pragma unroll
    for (int kt = 0; kt < 2; ++kt)
        #pragma unroll
        for (int e = 0; e < 16; ++e)
            ewB[kt][e] = sEW[kt * 32 + h * 16 + e];

    // biases / W4 in C layout: VGPR r of tile mt holds output unit o = mt*16 + 8h + r
    float b2r[4][8], b3r[2][8], w4r[2][8];
    #pragma unroll
    for (int mt = 0; mt < 4; ++mt)
        #pragma unroll
        for (int r = 0; r < 8; ++r)
            b2r[mt][r] = sB2[mt * 16 + h * 8 + r];
    #pragma unroll
    for (int mt = 0; mt < 2; ++mt)
        #pragma unroll
        for (int r = 0; r < 8; ++r) {
            b3r[mt][r] = sB3[mt * 16 + h * 8 + r];
            w4r[mt][r] = sW4v[mt * 16 + h * 8 + r];
        }
    const float b1f = b1[f];
    const float b4f = b4[f];

    _Float16* hsc = &sH2[w][0];
    const int chunk0 = blockIdx.x * ROWS_PER_CHUNK;

    for (int it = 0; it < ITERS; ++it) {
        const int b0 = chunk0 + it * ROWS_PER_ITER + w * TILE_ROWS;
        if (it + 1 < ITERS)
            __builtin_prefetch(&X[(b0 + ROWS_PER_ITER + m) * F_TOT + f], 0, 0);

        // ---- EXU layer: build h1^T B-operands directly in registers ----
        const float xc = X[(b0 + m) * F_TOT + f] - b1f;
        v16h Bh1[2];
        #pragma unroll
        for (int kt = 0; kt < 2; ++kt)
            #pragma unroll
            for (int e = 0; e < 16; ++e)
                Bh1[kt][e] = (_Float16)fminf(fmaxf(xc * ewB[kt][e], 0.0f), 1.0f);

        // ---- layer 2: h2^T = W2 x h1^T; packed b128 spill of h2[batch][o] ----
        #pragma unroll
        for (int mt = 0; mt < 4; ++mt) {
            v8f c = {};
            c = __builtin_amdgcn_wmma_f32_16x16x32_f16(false, Aw2[mt],     false, Bh1[0], (short)0, c, false, false);
            c = __builtin_amdgcn_wmma_f32_16x16x32_f16(false, Aw2[4 + mt], false, Bh1[1], (short)0, c, false, false);
            v8h hv;
            #pragma unroll
            for (int r = 0; r < 8; ++r)
                hv[r] = (_Float16)fmaxf(c[r] + b2r[mt][r], 0.0f);
            *(v8h*)&hsc[m * H2PAD + mt * 16 + h * 8] = hv; // 8 consecutive o values
        }
        asm volatile("s_wait_dscnt 0" ::: "memory"); // per-wave DS in-order; pin compiler ordering

        // ---- reload h2^T as layer-3 B-operands (contiguous 16-half slices) ----
        v16h Bh2[2];
        #pragma unroll
        for (int kt = 0; kt < 2; ++kt) {
            v8h lo = *(const v8h*)&hsc[m * H2PAD + kt * 32 + h * 16];
            v8h hi = *(const v8h*)&hsc[m * H2PAD + kt * 32 + h * 16 + 8];
            Bh2[kt] = __builtin_shufflevector(lo, hi, 0, 1, 2, 3, 4, 5, 6, 7,
                                                       8, 9, 10, 11, 12, 13, 14, 15);
        }

        // ---- layer 3 + fused bias/ReLU/W4 dot along the VGPR (o) axis ----
        float acc = 0.0f;
        #pragma unroll
        for (int mt = 0; mt < 2; ++mt) {
            v8f c = {};
            c = __builtin_amdgcn_wmma_f32_16x16x32_f16(false, Aw3[mt],     false, Bh2[0], (short)0, c, false, false);
            c = __builtin_amdgcn_wmma_f32_16x16x32_f16(false, Aw3[2 + mt], false, Bh2[1], (short)0, c, false, false);
            #pragma unroll
            for (int r = 0; r < 8; ++r)
                acc += fmaxf(c[r] + b3r[mt][r], 0.0f) * w4r[mt][r];
        }

        // lanes l and l^16 hold complementary o halves of the same batch row
        const float tot = acc + __shfl_xor(acc, 16, 32);
        if (h == 0)
            atomicAdd(&out[b0 + m], tot + b4f);
    }
}

extern "C" void kernel_launch(void* const* d_in, const int* in_sizes, int n_in,
                              void* d_out, int out_size, void* d_ws, size_t ws_size,
                              hipStream_t stream) {
    const float* X  = (const float*)d_in[0];
    const float* W1 = (const float*)d_in[1];
    const float* b1 = (const float*)d_in[2];
    const float* W2 = (const float*)d_in[3];
    const float* b2 = (const float*)d_in[4];
    const float* W3 = (const float*)d_in[5];
    const float* b3 = (const float*)d_in[6];
    const float* W4 = (const float*)d_in[7];
    const float* b4 = (const float*)d_in[8];
    const float* fb = (const float*)d_in[9];
    float* out = (float*)d_out;

    nam_init<<<(B_TOT + 255) / 256, 256, 0, stream>>>(out, fb);
    dim3 grid(CHUNKS, F_TOT);
    nam_fused<<<grid, 256, 0, stream>>>(X, W1, b1, W2, b2, W3, b3, W4, b4, out);
}
